// FFTConv_real_32615981646403
// MI455X (gfx1250) — matrix operand order repaired and checked by reading, not compiled, not run on hardware
//
#include <hip/hip_runtime.h>
#include <hip/hip_bf16.h>
#include <math.h>

// Problem constants (from the reference): B=32, S=524288, L=32768, CD=16.
#define NBATCH 32
#define SLEN   524288          // 2^19
#define LFFT   32768           // 2^15, filter length
#define BLOCK  16384           // L/2, overlap-add hop (= GEMM K)
#define CDIM   16
#define RPAD   64              // wrap padding for the reversed filter
#define GRLEN  (LFFT + RPAD)   // 32832 halves per reversed copy
#define CPY16  (GRLEN * 2 / 16)        // 16B chunks per LDS copy = 4104
#define LDSB   (2 * GRLEN * 2)         // dynamic LDS bytes = 131328

typedef __attribute__((ext_vector_type(16))) _Float16 v16h;
typedef __attribute__((ext_vector_type(8)))  _Float16 v8h;
typedef __attribute__((ext_vector_type(8)))  float    v8f;

// ---------------------------------------------------------------------------
// Kernel 1: mag[b,l] = sigmoid(c[b,:] . W[l,:] + bias[l]) -> f16, written as
// TWO reversed padded arrays:
//   gR [b][u] = mag_b[(-u)     & (L-1)],  u in [0, L+RPAD)
//   gRO[b][v] = mag_b[(-(v+1)) & (L-1)]  (= gR shifted by one element)
// so the circulant GEMM's gathers become contiguous ascending LDS runs with
// a parity-selectable 4-byte-aligned start.
// ---------------------------------------------------------------------------
__global__ __launch_bounds__(256)
void prep_mag_kernel(const float* __restrict__ c, const float* __restrict__ W,
                     const float* __restrict__ bias,
                     _Float16* __restrict__ gR, _Float16* __restrict__ gRO)
{
    int idx = blockIdx.x * 256 + threadIdx.x;          // idx = b*LFFT + l
    int b = idx >> 15;
    int l = idx & (LFFT - 1);
    const float* cb = c + b * CDIM;
    const float* wl = W + l * CDIM;
    float s = bias[l];
#pragma unroll
    for (int d = 0; d < CDIM; ++d) s = fmaf(cb[d], wl[d], s);
    _Float16 mh = (_Float16)(1.0f / (1.0f + __expf(-s)));

    _Float16* r  = gR  + (size_t)b * GRLEN;
    _Float16* ro = gRO + (size_t)b * GRLEN;
    int u = (LFFT - l) & (LFFT - 1);                   // (-l) mod L
    r[u] = mh;
    if (u < RPAD) r[u + LFFT] = mh;                    // periodic pad
    int v = LFFT - 1 - l;                              // (-(v+1)) mod L == l
    ro[v] = mh;
    if (v < RPAD) ro[v + LFFT] = mh;
}

// ---------------------------------------------------------------------------
// Kernel 2: xs[b,t] = (t >= L-1) ? inputs[b, t-(L-1)] : 0   -> f16
// (the F.pad(..., (L-1, 0)) left pad fused with the f16 downcast)
// ---------------------------------------------------------------------------
__global__ __launch_bounds__(256)
void prep_x_kernel(const float* __restrict__ in, _Float16* __restrict__ xs)
{
    int idx = blockIdx.x * 256 + threadIdx.x;          // idx = b*SLEN + t
    int t = idx & (SLEN - 1);
    xs[idx] = (t >= LFFT - 1) ? (_Float16)in[idx - (LFFT - 1)] : (_Float16)0.0f;
}

// ---------------------------------------------------------------------------
// Kernel 3: per-batch circulant GEMM + fused overlap-add scatter.
//   Y[b] (32 x 32768) = X[b] (32 x 16384) . G[b],  G[k][n] = mag_b[(n-k)&(L-1)]
//   y[b, blk*BLOCK + n] += Y[b][blk][n]  via global_atomic_add_f32.
// Grid (LFFT/256, NBATCH), 256 threads = 8 wave32s; wave w owns 32 columns
// x all 32 rows -> 4 f32 accumulator tiles -> 4 v_wmma_f32_16x16x32_f16/step.
// The two reversed filter copies (128 KB) are staged into LDS with
// global_load_async_to_lds_b128; B fragments are then 8 ds_load_b32 of a
// contiguous run already in v16h element order (no packing VALU).
// ---------------------------------------------------------------------------
__global__ __launch_bounds__(256)
void fftconv_circulant_wmma_kernel(const _Float16* __restrict__ xs,
                                   const _Float16* __restrict__ gR,
                                   const _Float16* __restrict__ gRO,
                                   float* __restrict__ out)
{
    extern __shared__ _Float16 smem[];                 // dynamic LDS, base off 0
    _Float16* sRR = smem;                              // RR[u]
    _Float16* sRO = smem + GRLEN;                      // RO[v] = RR[v+1]

    const int b       = blockIdx.y;
    const int n_tile0 = blockIdx.x * 256;
    const int tid     = threadIdx.x;

    // ---- async stage of both reversed copies into LDS (16B per lane per op)
    {
        const char* g0 = (const char*)(gR  + (size_t)b * GRLEN);
        const char* g1 = (const char*)(gRO + (size_t)b * GRLEN);
        for (int i = tid; i < CPY16; i += 256) {
            unsigned lo0 = (unsigned)(i * 16);                 // -> sRR
            unsigned lo1 = (unsigned)(GRLEN * 2 + i * 16);     // -> sRO
            const void* p0 = g0 + (size_t)i * 16;
            const void* p1 = g1 + (size_t)i * 16;
            asm volatile("global_load_async_to_lds_b128 %0, %1, off"
                         :: "v"(lo0), "v"(p0) : "memory");
            asm volatile("global_load_async_to_lds_b128 %0, %1, off"
                         :: "v"(lo1), "v"(p1) : "memory");
        }
        asm volatile("s_wait_asynccnt 0" ::: "memory");
    }
    __syncthreads();

    const int wave = tid >> 5;
    const int lane = tid & 31;
    const int l15  = lane & 15;
    const int hi   = lane >> 4;
    const int n0   = n_tile0 + wave * 32;              // this wave's 32 columns

    // Per-nt gather bases: element e of B frag = RR[(u0+k0)&M + e],
    // u0 = (hi*16 - n) & M; parity of the run start is k0-invariant, so the
    // RR/RO copy selection hoists out of the K loop.
    const int un0 = (hi * 16 - (n0 + l15))      & (LFFT - 1);
    const int un1 = (hi * 16 - (n0 + 16 + l15)) & (LFFT - 1);
    const _Float16* bb0 = (un0 & 1) ? (sRO - 1) : sRR;
    const _Float16* bb1 = (un1 & 1) ? (sRO - 1) : sRR;

    const _Float16* xb  = xs + (size_t)b * SLEN;
    const _Float16* ap0 = xb + (size_t)(l15)      * BLOCK + hi * 16;
    const _Float16* ap1 = xb + (size_t)(16 + l15) * BLOCK + hi * 16;

    v8f acc00 = {}, acc01 = {}, acc10 = {}, acc11 = {};

    union AU { v16h v; v8h h[2]; };
    union BU { v16h v; int  i[8]; };

#pragma unroll 2
    for (int k0 = 0; k0 < BLOCK; k0 += 32) {
        // ---- A fragments (16-bit A 16x32 layout): two b128 loads each,
        // landing directly in the 8 consecutive VGPRs of the operand.
        AU a0, a1;
        __builtin_prefetch((const void*)(ap0 + k0 + 256), 0, 1);
        a0.h[0] = *(const v8h*)(ap0 + k0);
        a0.h[1] = *(const v8h*)(ap0 + k0 + 8);
        a1.h[0] = *(const v8h*)(ap1 + k0);
        a1.h[1] = *(const v8h*)(ap1 + k0 + 8);

        // ---- B fragments: contiguous reversed-filter run, 8 x ds_load_b32.
        BU bf0, bf1;
        {
            const int* dp0 = (const int*)(bb0 + ((un0 + k0) & (LFFT - 1)));
            const int* dp1 = (const int*)(bb1 + ((un1 + k0) & (LFFT - 1)));
#pragma unroll
            for (int q = 0; q < 8; ++q) bf0.i[q] = dp0[q];
#pragma unroll
            for (int q = 0; q < 8; ++q) bf1.i[q] = dp1[q];
        }

        acc00 = __builtin_amdgcn_wmma_f32_16x16x32_f16(
                    false, a0.v, false, bf0.v, (short)0, acc00, false, false);
        acc10 = __builtin_amdgcn_wmma_f32_16x16x32_f16(
                    false, a1.v, false, bf0.v, (short)0, acc10, false, false);
        acc01 = __builtin_amdgcn_wmma_f32_16x16x32_f16(
                    false, a0.v, false, bf1.v, (short)0, acc01, false, false);
        acc11 = __builtin_amdgcn_wmma_f32_16x16x32_f16(
                    false, a1.v, false, bf1.v, (short)0, acc11, false, false);
    }

    // ---- Fused overlap-add scatter (f32 C layout: VGPR r <-> M=r+8*hi, lane=N).
    // Only (blk==31 && col>=BLOCK) falls past the S-sample truncation.
    float* ob = out + (size_t)b * SLEN;
#pragma unroll
    for (int mt = 0; mt < 2; ++mt) {
#pragma unroll
        for (int nt = 0; nt < 2; ++nt) {
            const v8f av = (mt == 0) ? (nt == 0 ? acc00 : acc01)
                                     : (nt == 0 ? acc10 : acc11);
            const int col  = n0 + nt * 16 + l15;
            const int rowb = mt * 16 + hi * 8;
#pragma unroll
            for (int r = 0; r < 8; ++r) {
                const int blk = rowb + r;              // overlap-add block index
                const int t   = blk * BLOCK + col;     // output sample
                if (blk < 31 || col < BLOCK) atomicAdd(ob + t, av[r]);
            }
        }
    }
}

// ---------------------------------------------------------------------------
// Host-side launch (graph-capture safe: only async ops on `stream`).
// Inputs: d_in[0]=inputs(B*S), d_in[1]=c(B*CD), d_in[2]=W(L*CD), d_in[3]=b(L)
// Workspace: gR | gRO (reversed padded filters, ~2.1MB each) | xs (32MB f16).
// ---------------------------------------------------------------------------
extern "C" void kernel_launch(void* const* d_in, const int* in_sizes, int n_in,
                              void* d_out, int out_size, void* d_ws, size_t ws_size,
                              hipStream_t stream)
{
    const float* in   = (const float*)d_in[0];
    const float* c    = (const float*)d_in[1];
    const float* W    = (const float*)d_in[2];
    const float* bias = (const float*)d_in[3];
    float*       out  = (float*)d_out;

    const size_t grBytes = (size_t)NBATCH * GRLEN * sizeof(_Float16);
    _Float16* gR  = (_Float16*)d_ws;
    _Float16* gRO = (_Float16*)((char*)d_ws + grBytes);
    _Float16* xs  = (_Float16*)((char*)d_ws + 2 * grBytes);

    // Output is accumulated with atomics -> zero it first (capture-safe).
    hipMemsetAsync(d_out, 0, (size_t)out_size * sizeof(float), stream);

    prep_mag_kernel<<<(NBATCH * LFFT) / 256, 256, 0, stream>>>(c, W, bias, gR, gRO);
    prep_x_kernel<<<(NBATCH * SLEN) / 256, 256, 0, stream>>>(in, xs);

    dim3 grid(LFFT / 256, NBATCH);
    fftconv_circulant_wmma_kernel<<<grid, 256, LDSB, stream>>>(xs, gR, gRO, out);
}
